// MambaDualClassifier_22213570855157
// MI455X (gfx1250) — compile-verified, matrix operand order checked by reading, not run on hardware
//
#include <hip/hip_runtime.h>
#include <hip/hip_bf16.h>
#include <math.h>

typedef __attribute__((ext_vector_type(16))) _Float16 v16h;
typedef __attribute__((ext_vector_type(8)))  _Float16 v8h;
typedef __attribute__((ext_vector_type(8)))  float    v8f;

#define D_MODEL 256
#define D_INNER 512
#define D_STATE 16
#define DT_RANK 16
#define D_CONV  4
#define HEAD_HID 128

__device__ __forceinline__ float silu_f(float v) {
    return v * (1.0f / (1.0f + __expf(-v)));
}
__device__ __forceinline__ float softplus_f(float v) {
    return (v > 20.0f) ? v : log1pf(__expf(v));
}

// ---------------------------------------------------------------------------
// Collapse the rank-1 input projection through in_proj:
//   c1[e] = sum_d in_proj_w[e,d] * proj_w[d]
//   c0[e] = sum_d in_proj_w[e,d] * proj_b[d]
// ---------------------------------------------------------------------------
__global__ void collapse_kernel(const float* __restrict__ inW,
                                const float* __restrict__ pw,
                                const float* __restrict__ pb,
                                float* __restrict__ c1,
                                float* __restrict__ c0) {
    int e = blockIdx.x * blockDim.x + threadIdx.x;
    if (e >= 2 * D_INNER) return;
    float s1 = 0.f, s0 = 0.f;
    const float* row = inW + (size_t)e * D_MODEL;
    for (int d = 0; d < D_MODEL; ++d) {
        float w = row[d];
        s1 += w * pw[d];
        s0 += w * pb[d];
    }
    c1[e] = s1;
    c0[e] = s0;
}

// ---------------------------------------------------------------------------
// Pack f32 weights into f16, optionally zero-padding K (Ksrc -> ldo):
//   out[n, k] = (k < Ksrc) ? f16(in[n, k]) : 0
// Used for x_proj_w (48x512) and dt_proj_w (512x16 -> 512x32 padded).
// ---------------------------------------------------------------------------
__global__ void pack_w16_kernel(const float* __restrict__ in, int ldi,
                                _Float16* __restrict__ out, int ldo,
                                int N, int Ksrc) {
    int idx = blockIdx.x * blockDim.x + threadIdx.x;
    if (idx >= N * ldo) return;
    int k = idx % ldo;
    int n = idx / ldo;
    out[idx] = (k < Ksrc) ? (_Float16)in[(size_t)n * ldi + k] : (_Float16)0.f;
}

// ---------------------------------------------------------------------------
// Causal depthwise conv1d + SiLU directly from the scalar input.
// Emits both f32 (for the scan) and f16 (GEMM operand) copies.
// ---------------------------------------------------------------------------
__global__ void conv_silu_kernel(const float* __restrict__ x,
                                 const float* __restrict__ c1,
                                 const float* __restrict__ c0,
                                 const float* __restrict__ cw,
                                 const float* __restrict__ cb,
                                 float* __restrict__ out,
                                 _Float16* __restrict__ out16,
                                 int B, int L) {
    int idx = blockIdx.x * blockDim.x + threadIdx.x;
    int total = B * L * D_INNER;
    if (idx >= total) return;
    int d = idx % D_INNER;
    int l = (idx / D_INNER) % L;
    int b = idx / (D_INNER * L);
    float k1 = c1[d], k0 = c0[d];
    float acc = cb[d];
#pragma unroll
    for (int j = 0; j < D_CONV; ++j) {
        int t = l - (D_CONV - 1) + j;
        float xin = (t >= 0) ? (x[b * L + t] * k1 + k0) : 0.f;
        acc += cw[d * D_CONV + j] * xin;
    }
    float s = silu_f(acc);
    out[idx]   = s;
    out16[idx] = (_Float16)s;
}

// ---------------------------------------------------------------------------
// f16 WMMA GEMM, guard-free:  out[m,n] = epi( sum_k A16[m,k] * W16[n,k] )
//   A16 row-major [M, lda] halves, W16 row-major [N, ldw] halves (K = mult 32).
//   One wave computes a 16 x (16*TN) tile: A chunk loaded once, TN WMMAs.
//   All operand loads are 16B-aligned v8h -> global_load_b128.
// Lane layout per CDNA5 ISA (16-bit A 16x32):
//   lanes 0-15: row = lane,    K in [kk,kk+8) and [kk+16,kk+24)
//   lanes 16-31: row = lane-16, K in [kk+8,kk+16) and [kk+24,kk+32)
// C/D: VGPR v -> row v + 8*(lane>=16), col = lane&15.
// EPI 0 (x_proj, TN==3, N==48): write f32 xdbc; cols 0-15 also written as f16
//   into dt16 [M,32] (cols 16-31 zero-padded by the t==1 sub-tile).
// EPI 1 (dt_proj): write f32 softplus(v + bias[n]).
// ---------------------------------------------------------------------------
template <int TN, int EPI>
__global__ void wmma_gemm16_kernel(const _Float16* __restrict__ A16, int lda,
                                   const _Float16* __restrict__ W16, int ldw,
                                   float* __restrict__ outF, int ldd,
                                   _Float16* __restrict__ dt16,
                                   const float* __restrict__ bias,
                                   int M, int N, int K) {
    int wavesPerBlock = blockDim.x >> 5;
    int waveId = threadIdx.x >> 5;
    int lane   = threadIdx.x & 31;
    int Mt = M >> 4;
    int Nt = N / (16 * TN);
    int tile = blockIdx.x * wavesPerBlock + waveId;
    if (tile >= Mt * Nt) return;          // uniform per wave: EXEC all-1s inside
    int tm = tile / Nt;
    int tn = tile % Nt;

    int r = lane & 15;
    int khalf = (lane >> 4) << 3;         // 0 or 8
    const _Float16* Arow = A16 + (size_t)(tm * 16 + r) * lda;
    const _Float16* Wrow[TN];
#pragma unroll
    for (int t = 0; t < TN; ++t)
        Wrow[t] = W16 + (size_t)(tn * 16 * TN + t * 16 + r) * ldw;

    v8f acc[TN];
#pragma unroll
    for (int t = 0; t < TN; ++t) acc[t] = (v8f){};

#pragma unroll 4
    for (int kk = 0; kk < K; kk += 32) {
        v8h alo = *(const v8h*)(Arow + kk + khalf);
        v8h ahi = *(const v8h*)(Arow + kk + khalf + 16);
        v16h av = __builtin_shufflevector(alo, ahi,
                    0, 1, 2, 3, 4, 5, 6, 7, 8, 9, 10, 11, 12, 13, 14, 15);
#pragma unroll
        for (int t = 0; t < TN; ++t) {
            v8h blo = *(const v8h*)(Wrow[t] + kk + khalf);
            v8h bhi = *(const v8h*)(Wrow[t] + kk + khalf + 16);
            v16h bv = __builtin_shufflevector(blo, bhi,
                        0, 1, 2, 3, 4, 5, 6, 7, 8, 9, 10, 11, 12, 13, 14, 15);
            acc[t] = __builtin_amdgcn_wmma_f32_16x16x32_f16(
                false, av, false, bv, (short)0, acc[t], false, false);
        }
    }

    int mbase = tm * 16 + khalf;          // rows 0-7 (lanes 0-15) or 8-15
#pragma unroll
    for (int t = 0; t < TN; ++t) {
        int ncol = tn * 16 * TN + t * 16 + r;
        float badd = (EPI == 1) ? bias[ncol] : 0.f;
#pragma unroll
        for (int v = 0; v < 8; ++v) {
            int m = mbase + v;
            float val = acc[t][v];
            if (EPI == 1) val = softplus_f(val + badd);
            outF[(size_t)m * ldd + ncol] = val;
            if (EPI == 0) {
                if (t == 0)      dt16[(size_t)m * 32 + ncol] = (_Float16)val;
                else if (t == 1) dt16[(size_t)m * 32 + ncol] = (_Float16)0.f;
            }
        }
    }
}

// ---------------------------------------------------------------------------
// Selective scan. One block per batch, one lane per inner channel d.
// h[16], A[16] in VGPRs; sequential over L; final (y + u*D)*silu(z) only at
// l = L-1 (the sole timestep the model consumes downstream).
// ---------------------------------------------------------------------------
__global__ void scan_kernel(const float* __restrict__ xc,    // [B,L,512]
                            const float* __restrict__ xdbc,  // [B,L,48]
                            const float* __restrict__ dt,    // [B,L,512]
                            const float* __restrict__ A_log, // [512,16]
                            const float* __restrict__ Dp,    // [512]
                            const float* __restrict__ x,     // [B,L]
                            const float* __restrict__ c1,    // [1024]
                            const float* __restrict__ c0,    // [1024]
                            float* __restrict__ yout,        // [B,512]
                            int L) {
    int b = blockIdx.x;
    int d = threadIdx.x;                  // 512 threads
    float Av[D_STATE];
#pragma unroll
    for (int n = 0; n < D_STATE; ++n) Av[n] = -__expf(A_log[d * D_STATE + n]);
    float h[D_STATE];
#pragma unroll
    for (int n = 0; n < D_STATE; ++n) h[n] = 0.f;

    const float* xcb = xc   + (size_t)b * L * D_INNER;
    const float* dtb = dt   + (size_t)b * L * D_INNER;
    const float* xpb = xdbc + (size_t)b * L * 48;

    for (int l = 0; l < L; ++l) {
        float dtv = dtb[(size_t)l * D_INNER + d];
        float u   = xcb[(size_t)l * D_INNER + d];
        if (l + 1 < L) {
            __builtin_prefetch(&dtb[(size_t)(l + 1) * D_INNER + d], 0, 0);
            __builtin_prefetch(&xcb[(size_t)(l + 1) * D_INNER + d], 0, 0);
        }
        float dtu = dtv * u;
        const float* Bc = xpb + (size_t)l * 48 + DT_RANK;
        const float* Cc = Bc + D_STATE;
#pragma unroll
        for (int n = 0; n < D_STATE; ++n) {
            float dA = __expf(dtv * Av[n]);
            h[n] = dA * h[n] + dtu * Bc[n];
        }
        if (l == L - 1) {
            float acc = 0.f;
#pragma unroll
            for (int n = 0; n < D_STATE; ++n) acc += h[n] * Cc[n];
            float yv = acc + u * Dp[d];
            float zv = x[b * L + (L - 1)] * c1[D_INNER + d] + c0[D_INNER + d];
            yout[b * D_INNER + d] = yv * silu_f(zv);
        }
    }
}

// ---------------------------------------------------------------------------
// out_proj at the final timestep only: out[b,o] = sum_e y[b,e]*W[o,e]
// ---------------------------------------------------------------------------
__global__ void outproj_kernel(const float* __restrict__ y,   // [8,512]
                               const float* __restrict__ W,   // [256,512]
                               float* __restrict__ out) {     // [8,256]
    int idx = blockIdx.x * blockDim.x + threadIdx.x;
    if (idx >= 8 * D_MODEL) return;
    int b = idx / D_MODEL, o = idx % D_MODEL;
    float s = 0.f;
    const float* yb = y + (size_t)b * D_INNER;
    const float* wr = W + (size_t)o * D_INNER;
    for (int e = 0; e < D_INNER; ++e) s += yb[e] * wr[e];
    out[idx] = s;
}

// ---------------------------------------------------------------------------
// Classifier head: relu(fused @ w1^T + b1) @ w2^T + b2. 8 blocks x 128 thr.
// ---------------------------------------------------------------------------
__global__ void head_kernel(const float* __restrict__ g,   // [8,256]
                            const float* __restrict__ l,   // [8,256]
                            const float* __restrict__ w1,  // [128,512]
                            const float* __restrict__ b1,  // [128]
                            const float* __restrict__ w2,  // [2,128]
                            const float* __restrict__ b2,  // [2]
                            float* __restrict__ out) {     // [8,2]
    int b = blockIdx.x;
    int j = threadIdx.x;
    __shared__ float hbuf[HEAD_HID];
    float s = b1[j];
    const float* wr = w1 + (size_t)j * (2 * D_MODEL);
    for (int i = 0; i < D_MODEL; ++i) s += g[b * D_MODEL + i] * wr[i];
    for (int i = 0; i < D_MODEL; ++i) s += l[b * D_MODEL + i] * wr[D_MODEL + i];
    hbuf[j] = fmaxf(s, 0.f);
    __syncthreads();
    if (j < 2) {
        float acc = b2[j];
        for (int k = 0; k < HEAD_HID; ++k) acc += hbuf[k] * w2[j * HEAD_HID + k];
        out[b * 2 + j] = acc;
    }
}

// ---------------------------------------------------------------------------
// Host side
// ---------------------------------------------------------------------------
struct MambaIn {
    const float *in_proj_w, *conv_w, *conv_b, *x_proj_w, *dt_proj_w,
                *dt_proj_b, *A_log, *Dvec, *out_proj_w;
};

static void run_branch(const float* x, int B, int L,
                       const float* proj_w, const float* proj_b,
                       const MambaIn& p,
                       float* c1, float* c0,
                       float* xc, _Float16* xc16,
                       _Float16* xw16, _Float16* dtw16,
                       float* xdbc, _Float16* dt16, float* dtbuf,
                       float* yfin, float* branch_out,
                       hipStream_t stream) {
    const int M = B * L;
    // 1. collapse rank-1 in_proj
    collapse_kernel<<<(2 * D_INNER + 255) / 256, 256, 0, stream>>>(
        p.in_proj_w, proj_w, proj_b, c1, c0);
    // 2. pack weights to f16 (x_proj 48x512; dt_proj 512x16 padded to 512x32)
    pack_w16_kernel<<<(48 * D_INNER + 255) / 256, 256, 0, stream>>>(
        p.x_proj_w, D_INNER, xw16, D_INNER, 48, D_INNER);
    pack_w16_kernel<<<(D_INNER * 32 + 255) / 256, 256, 0, stream>>>(
        p.dt_proj_w, DT_RANK, dtw16, 32, D_INNER, DT_RANK);
    // 3. conv + silu (f32 + f16 copies)
    int totalc = M * D_INNER;
    conv_silu_kernel<<<(totalc + 255) / 256, 256, 0, stream>>>(
        x, c1, c0, p.conv_w, p.conv_b, xc, xc16, B, L);
    // 4. x_proj WMMA: [M,512]f16 x [48,512]f16 -> xdbc[M,48]f32 + dt16[M,32]f16
    {
        int tiles = (M >> 4) * 1;                    // TN=3 covers all N=48
        wmma_gemm16_kernel<3, 0><<<(tiles + 7) / 8, 256, 0, stream>>>(
            xc16, D_INNER, xw16, D_INNER, xdbc, 48, dt16, nullptr,
            M, 48, D_INNER);
    }
    // 5. dt_proj WMMA (+bias+softplus): [M,32]f16 x [512,32]f16 -> dt[M,512]f32
    {
        int tiles = (M >> 4) * (D_INNER / 32);       // TN=2
        wmma_gemm16_kernel<2, 1><<<(tiles + 7) / 8, 256, 0, stream>>>(
            dt16, 32, dtw16, 32, dtbuf, D_INNER, nullptr, p.dt_proj_b,
            M, D_INNER, 32);
    }
    // 6. selective scan -> y * silu(z) at final step
    scan_kernel<<<B, D_INNER, 0, stream>>>(
        xc, xdbc, dtbuf, p.A_log, p.Dvec, x, c1, c0, yfin, L);
    // 7. out_proj on the final timestep
    outproj_kernel<<<(B * D_MODEL + 255) / 256, 256, 0, stream>>>(
        yfin, p.out_proj_w, branch_out);
}

extern "C" void kernel_launch(void* const* d_in, const int* in_sizes, int n_in,
                              void* d_out, int out_size, void* d_ws, size_t ws_size,
                              hipStream_t stream) {
    const float* x_global = (const float*)d_in[0];    // [8,2048]
    const float* x_local  = (const float*)d_in[1];    // [8,256]
    const float* gproj_w  = (const float*)d_in[2];
    const float* gproj_b  = (const float*)d_in[3];
    const float* lproj_w  = (const float*)d_in[4];
    const float* lproj_b  = (const float*)d_in[5];
    MambaIn mg = { (const float*)d_in[6],  (const float*)d_in[7],
                   (const float*)d_in[8],  (const float*)d_in[9],
                   (const float*)d_in[10], (const float*)d_in[11],
                   (const float*)d_in[12], (const float*)d_in[13],
                   (const float*)d_in[14] };
    MambaIn ml = { (const float*)d_in[15], (const float*)d_in[16],
                   (const float*)d_in[17], (const float*)d_in[18],
                   (const float*)d_in[19], (const float*)d_in[20],
                   (const float*)d_in[21], (const float*)d_in[22],
                   (const float*)d_in[23] };
    const float* head_w1 = (const float*)d_in[24];
    const float* head_b1 = (const float*)d_in[25];
    const float* head_w2 = (const float*)d_in[26];
    const float* head_b2 = (const float*)d_in[27];
    float* out = (float*)d_out;

    const int B = 8, Lg = 2048, Ll = 256;
    const int Mg = B * Lg, Mlc = B * Ll;

    // workspace carve-up, float units, each buffer aligned to 64 floats (256B)
    float* ws = (float*)d_ws;
    size_t off = 0;
    auto carve = [&](size_t nfloats) {
        float* p = ws + off;
        off += (nfloats + 63) & ~(size_t)63;
        return p;
    };
    float* c1g = carve(2 * D_INNER);
    float* c0g = carve(2 * D_INNER);
    float* c1l = carve(2 * D_INNER);
    float* c0l = carve(2 * D_INNER);
    float* xcg   = carve((size_t)Mg * D_INNER);
    float* xdbcg = carve((size_t)Mg * 48);
    float* dtg   = carve((size_t)Mg * D_INNER);
    float* xcl   = carve((size_t)Mlc * D_INNER);
    float* xdbcl = carve((size_t)Mlc * 48);
    float* dtl   = carve((size_t)Mlc * D_INNER);
    float* yg = carve((size_t)B * D_INNER);
    float* yl = carve((size_t)B * D_INNER);
    float* og = carve((size_t)B * D_MODEL);
    float* ol = carve((size_t)B * D_MODEL);
    // f16 buffers (half the float count, rounded up)
    _Float16* xc16g  = (_Float16*)carve((size_t)Mg * D_INNER / 2);
    _Float16* xc16l  = (_Float16*)carve((size_t)Mlc * D_INNER / 2);
    _Float16* dt16g  = (_Float16*)carve((size_t)Mg * 32 / 2);
    _Float16* dt16l  = (_Float16*)carve((size_t)Mlc * 32 / 2);
    _Float16* xw16g  = (_Float16*)carve((size_t)48 * D_INNER / 2);
    _Float16* xw16l  = (_Float16*)carve((size_t)48 * D_INNER / 2);
    _Float16* dtw16g = (_Float16*)carve((size_t)D_INNER * 32 / 2);
    _Float16* dtw16l = (_Float16*)carve((size_t)D_INNER * 32 / 2);

    run_branch(x_global, B, Lg, gproj_w, gproj_b, mg,
               c1g, c0g, xcg, xc16g, xw16g, dtw16g,
               xdbcg, dt16g, dtg, yg, og, stream);
    run_branch(x_local, B, Ll, lproj_w, lproj_b, ml,
               c1l, c0l, xcl, xc16l, xw16l, dtw16l,
               xdbcl, dt16l, dtl, yl, ol, stream);

    head_kernel<<<B, HEAD_HID, 0, stream>>>(
        og, ol, head_w1, head_b1, head_w2, head_b2, out);
}
